// MultiHeadAttention_14869176778893
// MI455X (gfx1250) — matrix operand order repaired and checked by reading, not compiled, not run on hardware
//
#include <hip/hip_runtime.h>

#define BB 4
#define SS 2048
#define DMODEL 1024
#define NH 16
#define HD 64

typedef __attribute__((ext_vector_type(16))) __bf16 v16bf;
typedef __attribute__((ext_vector_type(8)))  float  v8f;
typedef __attribute__((ext_vector_type(8)))  unsigned int v8u;

__device__ __forceinline__ __bf16 f2bf(float f) { return (__bf16)f; }  // native v_cvt, RNE

// Async DMA: global (16B per lane) -> LDS, tracked by ASYNCcnt.
// lds_byte_addr = low 32 bits of a generic pointer to __shared__ (the as3 offset);
// the LDS address is this VGPR plus the LDS base.
__device__ __forceinline__ void async_g2l_b128(unsigned lds_byte_addr, const void* gptr) {
  asm volatile("global_load_async_to_lds_b128 %0, %1, off"
               :
               : "v"(lds_byte_addr), "v"((unsigned long long)(size_t)gptr)
               : "memory");
}
__device__ __forceinline__ void wait_async0() {
  asm volatile("s_wait_asynccnt 0" ::: "memory");
}

// A-matrix 16x32 bf16 from row-major memory (row stride ld, elements).
// ISA layout: lanes 0-15 = row M, K=0..7 & 16..23; lanes 16-31 = row M, K=8..15 & 24..31.
__device__ __forceinline__ v16bf load_A16x32(const __bf16* p, int ld) {
  int lane = threadIdx.x & 31;
  int m  = lane & 15;
  int kb = (lane & 16) ? 8 : 0;
  const __bf16* row = p + (size_t)m * ld;
  uint4 r0 = *(const uint4*)(row + kb);        // K = kb .. kb+7
  uint4 r1 = *(const uint4*)(row + 16 + kb);   // K = 16+kb .. 16+kb+7
  v8u t = {r0.x, r0.y, r0.z, r0.w, r1.x, r1.y, r1.z, r1.w};
  return __builtin_bit_cast(v16bf, t);
}

// B-matrix 32x16 where B[k,n] = p[n*ld + k] (transposed-in-memory operand).
// ISA layout: lanes 0-15 = col N, K=0..7 & 16..23; lanes 16-31 = col N, K=8..15 & 24..31.
__device__ __forceinline__ v16bf load_BT16(const __bf16* p, int ld) {
  int lane = threadIdx.x & 31;
  int n  = lane & 15;
  int kb = (lane & 16) ? 8 : 0;
  const __bf16* row = p + (size_t)n * ld;
  uint4 r0 = *(const uint4*)(row + kb);
  uint4 r1 = *(const uint4*)(row + 16 + kb);
  v8u t = {r0.x, r0.y, r0.z, r0.w, r1.x, r1.y, r1.z, r1.w};
  return __builtin_bit_cast(v16bf, t);
}

__device__ __forceinline__ v8f wmma_bf16(v16bf a, v16bf b, v8f c) {
  return __builtin_amdgcn_wmma_f32_16x16x32_bf16(false, a, false, b, (short)0, c, false, false);
}

// ---------------- conversion f32 -> bf16 ----------------
__global__ void cvt_f32_bf16(const float* __restrict__ src, __bf16* __restrict__ dst, int n) {
  int i = blockIdx.x * blockDim.x + threadIdx.x;
  int stride = gridDim.x * blockDim.x;
  for (; i < n; i += stride) dst[i] = f2bf(src[i]);
}

// ---------------- conversion + transpose: WT[n*DM+k] = bf16(W[k*DM+n]) ----------------
__global__ void cvt_transpose(const float* __restrict__ src, __bf16* __restrict__ dst) {
  __shared__ float tile[16][17];
  int k0 = blockIdx.x * 16, n0 = blockIdx.y * 16;
  tile[threadIdx.y][threadIdx.x] = src[(size_t)(k0 + threadIdx.y) * DMODEL + n0 + threadIdx.x];
  __syncthreads();
  dst[(size_t)(n0 + threadIdx.y) * DMODEL + k0 + threadIdx.x] = f2bf(tile[threadIdx.x][threadIdx.y]);
}

// ---------------- fused QKV projection ----------------
// One wave computes a 32x64 tile of Q, K (row-major) or V (transposed to [b,h,d,s]).
__global__ __launch_bounds__(32) void qkv_proj(
    const __bf16* __restrict__ X,
    const __bf16* __restrict__ WqT, const __bf16* __restrict__ WkT, const __bf16* __restrict__ WvT,
    const float* __restrict__ bq, const float* __restrict__ bk, const float* __restrict__ bv,
    __bf16* __restrict__ Q, __bf16* __restrict__ K, __bf16* __restrict__ Vt) {
  int m0 = blockIdx.x * 32;
  int n0 = blockIdx.y * 64;
  int w  = blockIdx.z;
  const __bf16* WT   = (w == 0) ? WqT : (w == 1) ? WkT : WvT;
  const float*  bias = (w == 0) ? bq  : (w == 1) ? bk  : bv;

  v8f acc[2][4] = {};
  for (int k0 = 0; k0 < DMODEL; k0 += 32) {
    v16bf a0 = load_A16x32(X + (size_t)m0 * DMODEL + k0, DMODEL);
    v16bf a1 = load_A16x32(X + (size_t)(m0 + 16) * DMODEL + k0, DMODEL);
    __builtin_prefetch(X + (size_t)m0 * DMODEL + k0 + 32, 0, 0);
#pragma unroll
    for (int t = 0; t < 4; ++t) {
      const __bf16* bp = WT + (size_t)(n0 + 16 * t) * DMODEL + k0;
      v16bf b = load_BT16(bp, DMODEL);
      __builtin_prefetch(bp + 32, 0, 0);
      acc[0][t] = wmma_bf16(a0, b, acc[0][t]);
      acc[1][t] = wmma_bf16(a1, b, acc[1][t]);
    }
  }
  int lane  = threadIdx.x & 31;
  int n     = lane & 15;
  int rbase = (lane & 16) ? 8 : 0;
  if (w < 2) {
    __bf16* Out = (w == 0) ? Q : K;
#pragma unroll
    for (int mi = 0; mi < 2; ++mi)
#pragma unroll
      for (int t = 0; t < 4; ++t) {
        float bias_n = bias[n0 + 16 * t + n];
#pragma unroll
        for (int r = 0; r < 8; ++r) {
          Out[(size_t)(m0 + 16 * mi + rbase + r) * DMODEL + n0 + 16 * t + n] =
              f2bf(acc[mi][t][r] + bias_n);
        }
      }
  } else {
    // V stored transposed: Vt[((b*NH + h)*HD + d) * SS + s]
#pragma unroll
    for (int mi = 0; mi < 2; ++mi)
#pragma unroll
      for (int t = 0; t < 4; ++t) {
        int col = n0 + 16 * t + n;           // = h*HD + d
        int h   = col >> 6;
        int d   = col & 63;
        float bias_n = bias[col];
#pragma unroll
        for (int r = 0; r < 8; ++r) {
          int tok = m0 + 16 * mi + rbase + r;
          int b_  = tok >> 11;               // tok / SS
          int s_  = tok & (SS - 1);
          Vt[((size_t)(b_ * NH + h) * HD + d) * SS + s_] = f2bf(acc[mi][t][r] + bias_n);
        }
      }
  }
}

// ---------------- causal flash attention (cooperative, async-LDS-staged K/V) ----------------
// 4 waves / block; block covers a 64-query span (wave w owns queries q0s+16w .. +15).
// Each 64-key K/V tile is DMA'd to LDS via GLOBAL_LOAD_ASYNC_TO_LDS_B128 (ASYNCcnt),
// then consumed by all 4 waves via ds_load_b128 WMMA operand reads.
__global__ __launch_bounds__(128) void flash_attn(
    const __bf16* __restrict__ Q, const __bf16* __restrict__ K, const __bf16* __restrict__ Vt,
    __bf16* __restrict__ O) {
  __shared__ __align__(16) __bf16 Ktile[64 * 64];      // [key][d]
  __shared__ __align__(16) __bf16 Vtile[64 * 64];      // [d][key]
  __shared__ __align__(16) __bf16 Ptile[4][16 * 64];   // per-wave P
  int q0s  = blockIdx.x * 64;
  int h    = blockIdx.y;
  int b    = blockIdx.z;
  int tid  = threadIdx.x;
  int wid  = tid >> 5;
  int lane = tid & 31;
  int q0   = q0s + 16 * wid;
  int n    = lane & 15;
  int rbase = (lane & 16) ? 8 : 0;

  const __bf16* Kg = K + (size_t)b * SS * DMODEL + (size_t)h * HD;   // row s: stride DMODEL
  const __bf16* Vh = Vt + (size_t)(b * NH + h) * HD * SS;            // row d: stride SS
  __bf16* Pw = &Ptile[wid][0];
  const unsigned KtileB = (unsigned)(size_t)&Ktile[0];
  const unsigned VtileB = (unsigned)(size_t)&Vtile[0];

  const size_t baseQ = ((size_t)b * SS + q0) * DMODEL + (size_t)h * HD;
  v16bf aQ[2];
  aQ[0] = load_A16x32(Q + baseQ, DMODEL);
  aQ[1] = load_A16x32(Q + baseQ + 32, DMODEL);

  float mrow[8], lrow[8];
#pragma unroll
  for (int r = 0; r < 8; ++r) { mrow[r] = -3.0e38f; lrow[r] = 0.f; }
  v8f accO[4] = {};

  const int nkb = (q0s >> 6) + 1;  // key blocks 0 .. q0s step 64 (span causal limit)
  for (int ib = 0; ib < nkb; ++ib) {
    const int kb = ib * 64;
    __syncthreads();  // previous iteration's LDS reads complete
    // ---- async DMA: K (64 keys x 64 d) and V (64 d x 64 keys) into LDS ----
#pragma unroll
    for (int j = 0; j < 4; ++j) {
      int idx = j * 128 + tid;          // 0..511 16B chunks per tile
      int row = idx >> 3;
      int c   = (idx & 7) * 8;
      async_g2l_b128(KtileB + (unsigned)(row * 64 + c) * 2u,
                     Kg + (size_t)(kb + row) * DMODEL + c);
      async_g2l_b128(VtileB + (unsigned)(row * 64 + c) * 2u,
                     Vh + (size_t)row * SS + kb + c);
    }
    wait_async0();
    __syncthreads();

    // ---- logits: S = Q K^T / sqrt(D) (K^T operand read from LDS) ----
    v8f Sl[4] = {};
#pragma unroll
    for (int s = 0; s < 2; ++s) {
#pragma unroll
      for (int t = 0; t < 4; ++t) {
        v16bf bK = load_BT16(Ktile + (size_t)(16 * t) * 64 + 32 * s, 64);
        Sl[t] = wmma_bf16(aQ[s], bK, Sl[t]);
      }
    }
    float sf[8];
#pragma unroll
    for (int r = 0; r < 8; ++r) {
      int row = q0 + rbase + r;
      float mx = -3.0e38f;
#pragma unroll
      for (int t = 0; t < 4; ++t) {
        int col = kb + 16 * t + n;
        float v = Sl[t][r] * 0.125f;           // 1/sqrt(64)
        v = (col <= row) ? v : -10000.0f;      // causal mask, matches reference
        Sl[t][r] = v;
        mx = fmaxf(mx, v);
      }
      mx = fmaxf(mx, __shfl_xor(mx, 1, 32));
      mx = fmaxf(mx, __shfl_xor(mx, 2, 32));
      mx = fmaxf(mx, __shfl_xor(mx, 4, 32));
      mx = fmaxf(mx, __shfl_xor(mx, 8, 32));
      float mnew  = fmaxf(mrow[r], mx);
      float scale = __expf(mrow[r] - mnew);
      float rs = 0.f;
#pragma unroll
      for (int t = 0; t < 4; ++t) {
        float p = __expf(Sl[t][r] - mnew);
        rs += p;
        Pw[(rbase + r) * 64 + 16 * t + n] = f2bf(p);
      }
      rs += __shfl_xor(rs, 1, 32);
      rs += __shfl_xor(rs, 2, 32);
      rs += __shfl_xor(rs, 4, 32);
      rs += __shfl_xor(rs, 8, 32);
      lrow[r] = lrow[r] * scale + rs;
      mrow[r] = mnew;
      sf[r]   = scale;
    }
#pragma unroll
    for (int t = 0; t < 4; ++t)
#pragma unroll
      for (int r = 0; r < 8; ++r) accO[t][r] *= sf[r];

    // ---- O += P V (both operands from LDS; same-wave DS ordering is in-order) ----
#pragma unroll
    for (int s = 0; s < 2; ++s) {
      v16bf aP = load_A16x32(Pw + 32 * s, 64);
#pragma unroll
      for (int t = 0; t < 4; ++t) {
        v16bf bV = load_BT16(Vtile + (size_t)(16 * t) * 64 + 32 * s, 64);
        accO[t] = wmma_bf16(aP, bV, accO[t]);
      }
    }
  }
#pragma unroll
  for (int t = 0; t < 4; ++t)
#pragma unroll
    for (int r = 0; r < 8; ++r) {
      float o = accO[t][r] / lrow[r];
      O[((size_t)b * SS + q0 + rbase + r) * DMODEL + (size_t)h * HD + 16 * t + n] = f2bf(o);
    }
}

// ---------------- output projection + residual ----------------
__global__ __launch_bounds__(32) void out_proj(
    const __bf16* __restrict__ AO, const __bf16* __restrict__ WoT, const float* __restrict__ bo,
    const float* __restrict__ query, float* __restrict__ out) {
  int m0 = blockIdx.x * 32;
  int n0 = blockIdx.y * 64;
  v8f acc[2][4] = {};
  for (int k0 = 0; k0 < DMODEL; k0 += 32) {
    v16bf a0 = load_A16x32(AO + (size_t)m0 * DMODEL + k0, DMODEL);
    v16bf a1 = load_A16x32(AO + (size_t)(m0 + 16) * DMODEL + k0, DMODEL);
#pragma unroll
    for (int t = 0; t < 4; ++t) {
      const __bf16* bp = WoT + (size_t)(n0 + 16 * t) * DMODEL + k0;
      v16bf b = load_BT16(bp, DMODEL);
      __builtin_prefetch(bp + 32, 0, 0);
      acc[0][t] = wmma_bf16(a0, b, acc[0][t]);
      acc[1][t] = wmma_bf16(a1, b, acc[1][t]);
    }
  }
  int lane  = threadIdx.x & 31;
  int n     = lane & 15;
  int rbase = (lane & 16) ? 8 : 0;
#pragma unroll
  for (int mi = 0; mi < 2; ++mi)
#pragma unroll
    for (int t = 0; t < 4; ++t) {
      float bias_n = bo[n0 + 16 * t + n];
#pragma unroll
      for (int r = 0; r < 8; ++r) {
        size_t idx = (size_t)(m0 + 16 * mi + rbase + r) * DMODEL + n0 + 16 * t + n;
        out[idx] = acc[mi][t][r] + bias_n + query[idx];
      }
    }
}

// ---------------- row LayerNorm (in place on d_out) ----------------
__global__ __launch_bounds__(256) void layernorm_rows(
    float* __restrict__ out, const float* __restrict__ gamma, const float* __restrict__ beta) {
  __shared__ float red[256];
  int row = blockIdx.x;
  float* p = out + (size_t)row * DMODEL;
  int tid = threadIdx.x;
  float x[4];
  float s = 0.f;
#pragma unroll
  for (int i = 0; i < 4; ++i) { x[i] = p[tid + 256 * i]; s += x[i]; }
  red[tid] = s;
  __syncthreads();
  for (int off = 128; off > 0; off >>= 1) {
    if (tid < off) red[tid] += red[tid + off];
    __syncthreads();
  }
  float mu = red[0] / (float)DMODEL;
  __syncthreads();
  float vs = 0.f;
#pragma unroll
  for (int i = 0; i < 4; ++i) { float d = x[i] - mu; vs += d * d; }
  red[tid] = vs;
  __syncthreads();
  for (int off = 128; off > 0; off >>= 1) {
    if (tid < off) red[tid] += red[tid + off];
    __syncthreads();
  }
  float var  = red[0] / (float)DMODEL;
  float rstd = rsqrtf(var + 1e-6f);
#pragma unroll
  for (int i = 0; i < 4; ++i) {
    int c = tid + 256 * i;
    p[c] = (x[i] - mu) * rstd * gamma[c] + beta[c];
  }
}

extern "C" void kernel_launch(void* const* d_in, const int* in_sizes, int n_in,
                              void* d_out, int out_size, void* d_ws, size_t ws_size,
                              hipStream_t stream) {
  (void)in_sizes; (void)n_in; (void)out_size; (void)ws_size;
  const float* query = (const float*)d_in[0];
  // d_in[1] = mask: deterministic causal, applied analytically in-kernel.
  const float* Wq = (const float*)d_in[2];
  const float* bq = (const float*)d_in[3];
  const float* Wk = (const float*)d_in[4];
  const float* bk = (const float*)d_in[5];
  const float* Wv = (const float*)d_in[6];
  const float* bv = (const float*)d_in[7];
  const float* Wo = (const float*)d_in[8];
  const float* bo = (const float*)d_in[9];
  const float* gamma = (const float*)d_in[10];
  const float* beta  = (const float*)d_in[11];
  float* out = (float*)d_out;

  char* ws = (char*)d_ws;
  size_t off = 0;
  auto alloc = [&](size_t bytes) {
    void* p = ws + off;
    off += (bytes + 255) & ~(size_t)255;
    return p;
  };
  const size_t NTOK = (size_t)BB * SS;  // 8192 rows
  __bf16* Xb   = (__bf16*)alloc(NTOK * DMODEL * 2);
  __bf16* WqTb = (__bf16*)alloc((size_t)DMODEL * DMODEL * 2);
  __bf16* WkTb = (__bf16*)alloc((size_t)DMODEL * DMODEL * 2);
  __bf16* WvTb = (__bf16*)alloc((size_t)DMODEL * DMODEL * 2);
  __bf16* WoTb = (__bf16*)alloc((size_t)DMODEL * DMODEL * 2);
  __bf16* Qb   = (__bf16*)alloc(NTOK * DMODEL * 2);
  __bf16* Kb   = (__bf16*)alloc(NTOK * DMODEL * 2);
  __bf16* Vtb  = (__bf16*)alloc(NTOK * DMODEL * 2);
  __bf16* AOb  = (__bf16*)alloc(NTOK * DMODEL * 2);

  cvt_f32_bf16<<<2048, 256, 0, stream>>>(query, Xb, (int)(NTOK * DMODEL));
  dim3 gt(DMODEL / 16, DMODEL / 16);
  dim3 bt(16, 16);
  cvt_transpose<<<gt, bt, 0, stream>>>(Wq, WqTb);
  cvt_transpose<<<gt, bt, 0, stream>>>(Wk, WkTb);
  cvt_transpose<<<gt, bt, 0, stream>>>(Wv, WvTb);
  cvt_transpose<<<gt, bt, 0, stream>>>(Wo, WoTb);

  dim3 gq((unsigned)(NTOK / 32), DMODEL / 64, 3);
  qkv_proj<<<gq, 32, 0, stream>>>(Xb, WqTb, WkTb, WvTb, bq, bk, bv, Qb, Kb, Vtb);

  dim3 ga(SS / 64, NH, BB);
  flash_attn<<<ga, 128, 0, stream>>>(Qb, Kb, Vtb, AOb);

  dim3 gp((unsigned)(NTOK / 32), DMODEL / 64, 1);
  out_proj<<<gp, 32, 0, stream>>>(AOb, WoTb, bo, query, out);

  layernorm_rows<<<(unsigned)NTOK, 256, 0, stream>>>(out, gamma, beta);
}
